// MDN_55087250538537
// MI455X (gfx1250) — matrix operand order repaired and checked by reading, not compiled, not run on hardware
//
#include <hip/hip_runtime.h>
#include <math.h>

// Problem dims (fixed by the reference)
#define BDIM 4096
#define INDIM 512
#define HDIM 1024
#define ODIM 256
#define GDIM 3
#define H3   (3 * HDIM)
#define GO   (GDIM * ODIM)   // 768

typedef __bf16 bf16_t;
typedef __bf16 v16bf __attribute__((ext_vector_type(16)));
typedef float  v8f   __attribute__((ext_vector_type(8)));

// LDS tile row stride in halves: 32 data + 8 pad = 80B rows (16B aligned,
// 20-bank rotation per row -> conflict-free b128 fragment loads).
#define LDS_STRIDE 40

__device__ __forceinline__ v8f wmma_bf16(v16bf a, v16bf b, v8f c) {
  // D = A(16x32) x B(32x16) + C, f32 accumulate
  return __builtin_amdgcn_wmma_f32_16x16x32_bf16(
      /*neg_a=*/false, a, /*neg_b=*/false, b,
      /*c_mod=*/(short)0, c, /*reuse_a=*/false, /*reuse_b=*/false);
}

union FragBF { v16bf v; uint4 q[2]; };

// A fragment: rows [rbase, rbase+16) of an LDS tile, 32 K-halves per row.
// lane: m = lane&15 ; hi = lane>>4 selects K-subgroup per ISA A-layout.
__device__ __forceinline__ v16bf load_frag_a(const bf16_t* tile, int rbase, int lane) {
  const int m = lane & 15, hi = lane >> 4;
  const bf16_t* p = tile + (rbase + m) * LDS_STRIDE + hi * 8;
  FragBF f;
  f.q[0] = *(const uint4*)(p);        // K = hi*8 .. hi*8+7      (VGPR 0-3)
  f.q[1] = *(const uint4*)(p + 16);   // K = 16 + hi*8 .. +7     (VGPR 4-7)
  return f.v;
}

// B fragment: weight rows (= output cols) [cbase, cbase+16).
// lane: n = lane&15 ; khi = lane>>4 selects K-subgroup (K=khi*8..+7, 16+khi*8..+7).
__device__ __forceinline__ v16bf load_frag_b(const bf16_t* tile, int cbase, int lane) {
  const int n = lane & 15, khi = lane >> 4;
  const bf16_t* p = tile + (cbase + n) * LDS_STRIDE + khi * 8;
  FragBF f;
  f.q[0] = *(const uint4*)(p);
  f.q[1] = *(const uint4*)(p + 16);
  return f.v;
}

__device__ __forceinline__ float sigmoidf_(float x) { return 1.0f / (1.0f + __expf(-x)); }
__device__ __forceinline__ float softplusf_(float x) { return x > 20.0f ? x : log1pf(__expf(x)); }

// ---------------------------------------------------------------- convert
__global__ __launch_bounds__(256) void cvt_f32_bf16(const float* __restrict__ src,
                                                    bf16_t* __restrict__ dst, int n) {
  int i = blockIdx.x * blockDim.x + threadIdx.x;
  const int stride = gridDim.x * blockDim.x;
  for (; i < n; i += stride) dst[i] = (bf16_t)src[i];
}

// ---------------------------------------------------------------- h1 = relu(x Wt + b)
// grid (B/64, H/64), 256 threads = 8 waves: 4 row-waves x 2 col-waves (2 tiles each)
__global__ __launch_bounds__(256) void i2d_relu_kernel(
    const bf16_t* __restrict__ x_bf,     // [B][IN]
    const bf16_t* __restrict__ w_bf,     // [H][IN]
    const float*  __restrict__ bias,     // [H]
    bf16_t* __restrict__ h1_bf) {        // [B][H]
  __shared__ bf16_t sA[64 * LDS_STRIDE];
  __shared__ bf16_t sW[64 * LDS_STRIDE];
  const int t = threadIdx.x, lane = t & 31, wid = t >> 5;
  const int wr = wid & 3, wc = wid >> 2;
  const int bm0 = blockIdx.x * 64, n0 = blockIdx.y * 64;

  v8f acc[2] = {};
  for (int k0 = 0; k0 < INDIM; k0 += 32) {
    const int row = t >> 2, ch = t & 3;
    *(uint4*)&sA[row * LDS_STRIDE + ch * 8] =
        *(const uint4*)&x_bf[(size_t)(bm0 + row) * INDIM + k0 + ch * 8];
    *(uint4*)&sW[row * LDS_STRIDE + ch * 8] =
        *(const uint4*)&w_bf[(size_t)(n0 + row) * INDIM + k0 + ch * 8];
    __syncthreads();
    const v16bf fa = load_frag_a(sA, wr * 16, lane);
#pragma unroll
    for (int c = 0; c < 2; ++c) {
      const v16bf fb = load_frag_b(sW, wc * 32 + c * 16, lane);
      acc[c] = wmma_bf16(fa, fb, acc[c]);
    }
    __syncthreads();
  }
#pragma unroll
  for (int c = 0; c < 2; ++c) {
    const int n = n0 + wc * 32 + c * 16 + (lane & 15);
    const float b = bias[n];
#pragma unroll
    for (int v = 0; v < 8; ++v) {
      const int m = bm0 + wr * 16 + v + 8 * (lane >> 4);
      float val = acc[c][v] + b;
      val = val > 0.f ? val : 0.f;
      h1_bf[(size_t)m * HDIM + n] = (bf16_t)val;
    }
  }
}

// ---------------------------------------------------------------- fused GRU cell
// One workgroup computes a 64x64 tile of h_new; each wave holds 6 accumulators
// (r,z,n) x (ih,hh) per 16x16 subtile, gate math done in-register.
__global__ __launch_bounds__(256) void gru_cell_kernel(
    const bf16_t* __restrict__ inp_bf,   // [B][H]
    const bf16_t* __restrict__ hp_bf,    // [B][H]
    const float*  __restrict__ hp_f32,   // [B][H]
    const bf16_t* __restrict__ wih,      // [3H][H] rows: r,z,n
    const bf16_t* __restrict__ whh,      // [3H][H]
    const float*  __restrict__ bih,      // [3H]
    const float*  __restrict__ bhh,      // [3H]
    float*  __restrict__ h_out_f32,      // [B][H] -> hidden_out slice of d_out
    bf16_t* __restrict__ next_bf,        // [B][H] next-layer input (maybe tanh'd)
    int apply_tanh) {
  __shared__ bf16_t sI[64 * LDS_STRIDE];
  __shared__ bf16_t sH[64 * LDS_STRIDE];
  __shared__ bf16_t sW[6][64 * LDS_STRIDE];   // [ih_r, ih_z, ih_n, hh_r, hh_z, hh_n]
  const int t = threadIdx.x, lane = t & 31, wid = t >> 5;
  const int wr = wid & 3, wc = wid >> 2;
  const int bm0 = blockIdx.x * 64, n0 = blockIdx.y * 64;

  v8f acc[2][6] = {};
  for (int k0 = 0; k0 < HDIM; k0 += 32) {
    {
      const int row = t >> 2, ch = t & 3;
      *(uint4*)&sI[row * LDS_STRIDE + ch * 8] =
          *(const uint4*)&inp_bf[(size_t)(bm0 + row) * HDIM + k0 + ch * 8];
      *(uint4*)&sH[row * LDS_STRIDE + ch * 8] =
          *(const uint4*)&hp_bf[(size_t)(bm0 + row) * HDIM + k0 + ch * 8];
    }
#pragma unroll
    for (int idx = t; idx < 1536; idx += 256) {       // 6 gate tiles x 64 rows x 4 chunks
      const int g = idx >> 8, rem = idx & 255;
      const int row = rem >> 2, ch = rem & 3;
      const bf16_t* W = (g < 3) ? wih : whh;
      const int gate = g - ((g < 3) ? 0 : 3);
      *(uint4*)&sW[g][row * LDS_STRIDE + ch * 8] =
          *(const uint4*)&W[(size_t)(gate * HDIM + n0 + row) * HDIM + k0 + ch * 8];
    }
    __syncthreads();
    const v16bf fi = load_frag_a(sI, wr * 16, lane);
    const v16bf fh = load_frag_a(sH, wr * 16, lane);
#pragma unroll
    for (int c = 0; c < 2; ++c) {
      const int cb = wc * 32 + c * 16;
#pragma unroll
      for (int g = 0; g < 3; ++g) {
        acc[c][g]     = wmma_bf16(fi, load_frag_b(sW[g],     cb, lane), acc[c][g]);
        acc[c][3 + g] = wmma_bf16(fh, load_frag_b(sW[3 + g], cb, lane), acc[c][3 + g]);
      }
    }
    __syncthreads();
  }
  // gate math + blend, all in-register (PyTorch cell: r,z,n order)
#pragma unroll
  for (int c = 0; c < 2; ++c) {
    const int n = n0 + wc * 32 + c * 16 + (lane & 15);
    const float bir = bih[n], biz = bih[HDIM + n], bin = bih[2 * HDIM + n];
    const float bhr = bhh[n], bhz = bhh[HDIM + n], bhn = bhh[2 * HDIM + n];
#pragma unroll
    for (int v = 0; v < 8; ++v) {
      const int m = bm0 + wr * 16 + v + 8 * (lane >> 4);
      const float r  = sigmoidf_((acc[c][0][v] + bir) + (acc[c][3][v] + bhr));
      const float z  = sigmoidf_((acc[c][1][v] + biz) + (acc[c][4][v] + bhz));
      const float nn = tanhf((acc[c][2][v] + bin) + r * (acc[c][5][v] + bhn));
      const float hp = hp_f32[(size_t)m * HDIM + n];
      const float hn = (1.0f - z) * nn + z * hp;
      h_out_f32[(size_t)m * HDIM + n] = hn;
      next_bf[(size_t)m * HDIM + n] = (bf16_t)(apply_tanh ? tanhf(hn) : hn);
    }
  }
}

// ---------------------------------------------------------------- fused MDN heads
// grid (B/128, O/16); 8 waves x 16 rows; each wave: 9 accs = (mu,sig,pi) x 3 groups
// so pi's group-softmax and sigma's softplus happen in-register.
__global__ __launch_bounds__(256) void mdn_head_kernel(
    const bf16_t* __restrict__ hout_bf,  // [B][H] (= tanh of last GRU state)
    const bf16_t* __restrict__ headw,    // [3][GO][H]: mu, sigma, pi stacked
    const float* __restrict__ mu_b, const float* __restrict__ sg_b,
    const float* __restrict__ pi_b,
    float* __restrict__ out_mu, float* __restrict__ out_sg,
    float* __restrict__ out_pi) {        // each [B][G][O]
  __shared__ bf16_t sA[128 * LDS_STRIDE];
  __shared__ bf16_t sW[9][16 * LDS_STRIDE];
  const int t = threadIdx.x, lane = t & 31, wid = t >> 5;
  const int bm0 = blockIdx.x * 128, o0 = blockIdx.y * 16;

  v8f acc[9] = {};
  for (int k0 = 0; k0 < HDIM; k0 += 32) {
#pragma unroll
    for (int idx = t; idx < 512; idx += 256) {        // 128 rows x 4 chunks
      const int row = idx >> 2, ch = idx & 3;
      *(uint4*)&sA[row * LDS_STRIDE + ch * 8] =
          *(const uint4*)&hout_bf[(size_t)(bm0 + row) * HDIM + k0 + ch * 8];
    }
    for (int idx = t; idx < 576; idx += 256) {        // 9 tiles x 16 rows x 4 chunks
      const int g = idx >> 6, rem = idx & 63;
      const int row = rem >> 2, ch = rem & 3;
      const int head = g / 3, grp = g % 3;
      *(uint4*)&sW[g][row * LDS_STRIDE + ch * 8] =
          *(const uint4*)&headw[(size_t)(head * GO + grp * ODIM + o0 + row) * HDIM + k0 + ch * 8];
    }
    __syncthreads();
    const v16bf fa = load_frag_a(sA, wid * 16, lane);
#pragma unroll
    for (int g = 0; g < 9; ++g)
      acc[g] = wmma_bf16(fa, load_frag_b(sW[g], 0, lane), acc[g]);
    __syncthreads();
  }
  const int o = o0 + (lane & 15);
  const float bm0f = mu_b[o], bm1f = mu_b[ODIM + o], bm2f = mu_b[2 * ODIM + o];
  const float bs0 = sg_b[o], bs1 = sg_b[ODIM + o], bs2 = sg_b[2 * ODIM + o];
  const float bp0 = pi_b[o], bp1 = pi_b[ODIM + o], bp2 = pi_b[2 * ODIM + o];
#pragma unroll
  for (int v = 0; v < 8; ++v) {
    const int m = bm0 + wid * 16 + v + 8 * (lane >> 4);
    const size_t base = (size_t)m * GO + o;
    // pi: softmax over the 3 groups
    const float p0 = acc[6][v] + bp0, p1 = acc[7][v] + bp1, p2 = acc[8][v] + bp2;
    const float mx = fmaxf(p0, fmaxf(p1, p2));
    const float e0 = __expf(p0 - mx), e1 = __expf(p1 - mx), e2 = __expf(p2 - mx);
    const float inv = 1.0f / (e0 + e1 + e2);
    out_mu[base]            = acc[0][v] + bm0f;
    out_mu[base + ODIM]     = acc[1][v] + bm1f;
    out_mu[base + 2 * ODIM] = acc[2][v] + bm2f;
    out_sg[base]            = softplusf_(acc[3][v] + bs0);
    out_sg[base + ODIM]     = softplusf_(acc[4][v] + bs1);
    out_sg[base + 2 * ODIM] = softplusf_(acc[5][v] + bs2);
    out_pi[base]            = e0 * inv;
    out_pi[base + ODIM]     = e1 * inv;
    out_pi[base + 2 * ODIM] = e2 * inv;
  }
}

// ---------------------------------------------------------------- launcher
extern "C" void kernel_launch(void* const* d_in, const int* in_sizes, int n_in,
                              void* d_out, int out_size, void* d_ws, size_t ws_size,
                              hipStream_t stream) {
  (void)in_sizes; (void)n_in; (void)out_size; (void)ws_size;
  const float* x      = (const float*)d_in[0];   // [B,IN]
  const float* hidden = (const float*)d_in[1];   // [L,B,H]
  const float* i2d_w  = (const float*)d_in[2];   // [H,IN]
  const float* i2d_b  = (const float*)d_in[3];   // [H]
  const float* gwih   = (const float*)d_in[4];   // [L,3H,H]
  const float* gwhh   = (const float*)d_in[5];   // [L,3H,H]
  const float* gbih   = (const float*)d_in[6];   // [L,3H]
  const float* gbhh   = (const float*)d_in[7];   // [L,3H]
  const float* mu_w   = (const float*)d_in[8];   // [GO,H]
  const float* mu_b   = (const float*)d_in[9];
  const float* sg_w   = (const float*)d_in[10];
  const float* sg_b   = (const float*)d_in[11];
  const float* pi_w   = (const float*)d_in[12];
  const float* pi_b   = (const float*)d_in[13];
  float* out = (float*)d_out;  // [mu | sig | pi | hidden_out]

  char* ws = (char*)d_ws;
  size_t off = 0;
  auto walloc = [&](size_t bytes) -> char* {
    char* p = ws + off;
    off += (bytes + 255) & ~(size_t)255;
    return p;
  };
  bf16_t* x_bf    = (bf16_t*)walloc((size_t)BDIM * INDIM * 2);
  bf16_t* hid_bf  = (bf16_t*)walloc((size_t)2 * BDIM * HDIM * 2);
  bf16_t* i2dw_bf = (bf16_t*)walloc((size_t)HDIM * INDIM * 2);
  bf16_t* wih_bf  = (bf16_t*)walloc((size_t)2 * H3 * HDIM * 2);
  bf16_t* whh_bf  = (bf16_t*)walloc((size_t)2 * H3 * HDIM * 2);
  bf16_t* headw_bf= (bf16_t*)walloc((size_t)3 * GO * HDIM * 2);
  bf16_t* h1_bf   = (bf16_t*)walloc((size_t)BDIM * HDIM * 2);
  bf16_t* a0_bf   = (bf16_t*)walloc((size_t)BDIM * HDIM * 2);
  bf16_t* ho_bf   = (bf16_t*)walloc((size_t)BDIM * HDIM * 2);

  auto cvt = [&](const float* s, bf16_t* d, long n) {
    cvt_f32_bf16<<<2048, 256, 0, stream>>>(s, d, (int)n);
  };
  cvt(x,      x_bf,    (long)BDIM * INDIM);
  cvt(hidden, hid_bf,  (long)2 * BDIM * HDIM);
  cvt(i2d_w,  i2dw_bf, (long)HDIM * INDIM);
  cvt(gwih,   wih_bf,  (long)2 * H3 * HDIM);
  cvt(gwhh,   whh_bf,  (long)2 * H3 * HDIM);
  cvt(mu_w,   headw_bf,                     (long)GO * HDIM);
  cvt(sg_w,   headw_bf + (size_t)GO * HDIM, (long)GO * HDIM);
  cvt(pi_w,   headw_bf + (size_t)2 * GO * HDIM, (long)GO * HDIM);

  // h1 = relu(x W1^T + b)
  i2d_relu_kernel<<<dim3(BDIM / 64, HDIM / 64), 256, 0, stream>>>(
      x_bf, i2dw_bf, i2d_b, h1_bf);

  float* hid_out = out + (size_t)3 * BDIM * GO;  // hidden_out region [L,B,H]

  // GRU layer 0
  gru_cell_kernel<<<dim3(BDIM / 64, HDIM / 64), 256, 0, stream>>>(
      h1_bf, hid_bf, hidden, wih_bf, whh_bf, gbih, gbhh,
      hid_out, a0_bf, /*apply_tanh=*/0);
  // GRU layer 1 (its bf16 side-output is tanh(h) = head input)
  gru_cell_kernel<<<dim3(BDIM / 64, HDIM / 64), 256, 0, stream>>>(
      a0_bf, hid_bf + (size_t)BDIM * HDIM, hidden + (size_t)BDIM * HDIM,
      wih_bf + (size_t)H3 * HDIM, whh_bf + (size_t)H3 * HDIM,
      gbih + H3, gbhh + H3,
      hid_out + (size_t)BDIM * HDIM, ho_bf, /*apply_tanh=*/1);

  // heads: mu / softplus(sigma) / softmax(pi over groups)
  mdn_head_kernel<<<dim3(BDIM / 128, ODIM / 16), 256, 0, stream>>>(
      ho_bf, headw_bf, mu_b, sg_b, pi_b,
      out, out + (size_t)BDIM * GO, out + (size_t)2 * BDIM * GO);
}
